// attention_2851858285082
// MI455X (gfx1250) — matrix operand order repaired and checked by reading, not compiled, not run on hardware
//
#include <hip/hip_runtime.h>
#include <hip/hip_bf16.h>

// Problem constants (from reference): B=4, N=2048, D=1024, H=16, Dh=64
#define B_  4
#define N_  2048
#define D_  1024
#define H_  16
#define DH_ 64

typedef _Float16 half8 __attribute__((ext_vector_type(8)));
typedef _Float16 v16h  __attribute__((ext_vector_type(16)));
typedef float    v8f   __attribute__((ext_vector_type(8)));

__device__ __forceinline__ v16h hcat(half8 lo, half8 hi) {
  v16h r;
#pragma unroll
  for (int i = 0; i < 8; ++i) { r[i] = lo[i]; r[i + 8] = hi[i]; }
  return r;
}

// A-operand fragment (16x32 f16).  ISA layout: lane L (m=L&15, g=L>>4) holds
// row m; half h maps to K = (h&8)*2 + g*8 + (h&7)  -> two contiguous 16B runs.
__device__ __forceinline__ v16h load_fragA(const _Float16* p, int g, int k0) {
  half8 lo = *(const half8*)(p + k0 + g * 8);
  half8 hi = *(const half8*)(p + k0 + 16 + g * 8);
  return hcat(lo, hi);
}

// B-operand fragment (32x16 f16) from K-contiguous storage (column-major tile).
// Lane L holds column n=L&15; half h maps to K = (h&8)*2 + g*8 + (h&7) -> two 16B runs.
__device__ __forceinline__ v16h load_fragB(const _Float16* pn, int g, int k0) {
  half8 lo = *(const half8*)(pn + k0 + g * 8);
  half8 hi = *(const half8*)(pn + k0 + 16 + g * 8);
  return hcat(lo, hi);
}

__device__ __forceinline__ v8f wmma32(v16h a, v16h b, v8f c) {
  return __builtin_amdgcn_wmma_f32_16x16x32_f16(false, a, false, b, (short)0, c,
                                                false, false);
}

// ---------------------------------------------------------------------------
// Tensor Data Mover: 2D f16 tile load, global -> LDS, with LDS row padding.
// Descriptor per CDNA5 ISA 8.3/8.4.  This toolchain exposes the 6-arg builtin:
//   (u32x4 g0, i32x8 g1, i32x4 g2, i32x4 g3, i32x8 extra, i32 cpol)
// ---------------------------------------------------------------------------
#if __has_builtin(__builtin_amdgcn_tensor_load_to_lds)
#define HAVE_TDM 1
typedef unsigned int u32x4 __attribute__((ext_vector_type(4)));
typedef int          i32x8 __attribute__((ext_vector_type(8)));
typedef int          i32x4 __attribute__((ext_vector_type(4)));

__device__ __forceinline__ void tdm_load_2d_f16(
    unsigned int lds_addr, const _Float16* gptr, unsigned int pad_ctrl,
    unsigned int tensor_d0, unsigned int tensor_d1, unsigned int tile_d0,
    unsigned int tile_d1, unsigned int stride0) {
  unsigned long long ga = (unsigned long long)(uintptr_t)gptr;
  u32x4 g0;
  g0[0] = 1u;                    // count=1, user descriptor, no gather
  g0[1] = lds_addr;              // LDS byte address (flat-ptr low 32 bits)
  g0[2] = (unsigned int)ga;      // global_addr[31:0]
  g0[3] = ((unsigned int)(ga >> 32) & 0x01ffffffu) | (2u << 30);  // [56:32]|type=2
  i32x8 g1;
  g1[0] = (int)(0x00010000u | pad_ctrl);  // data_size=1 (2B) | pad config
  g1[1] = (int)((tensor_d0 & 0xffffu) << 16);                    // dim0[15:0]
  g1[2] = (int)((tensor_d0 >> 16) | ((tensor_d1 & 0xffffu) << 16));
  g1[3] = (int)((tensor_d1 >> 16) | (tile_d0 << 16));
  g1[4] = (int)tile_d1;                   // tile_dim1 (tile_dim2 = 0)
  g1[5] = (int)stride0;                   // tensor_dim0_stride[31:0]
  g1[6] = 0;
  g1[7] = 0;
  i32x4 z4 = {0, 0, 0, 0};                // groups 2/3 unused (2D tensor)
  i32x8 z8 = {0, 0, 0, 0, 0, 0, 0, 0};
  __builtin_amdgcn_tensor_load_to_lds(g0, g1, z4, z4, z8, 0);
}
// pad_enable | pad_interval | pad_amount(=4 dwords -> +8 halves per row)
#define PAD_ROW64H ((1u << 20) | (4u << 22) | (3u << 25))  // 32-dword rows -> stride 72
#define PAD_ROW32H ((1u << 20) | (3u << 22) | (3u << 25))  // 16-dword rows -> stride 40
#else
#define HAVE_TDM 0
#endif

// ---------------------------------------------------------------------------
// Kernel 1: fused QKV projection GEMM + bias + RoPE.
//   Q,K -> f16 [B,H,N,Dh] (RoPE applied);  V -> f16 [B,H,Dh,N] (transposed so
//   the attention kernel's TDM tile load lands in PV B-fragment layout).
// grid = (M/128, D/64, 3), block = 256 (8 waves, 4x2 wave grid, 32x32/wave)
// ---------------------------------------------------------------------------
__global__ __launch_bounds__(256) void qkv_rope_gemm(
    const float* __restrict__ x,
    const float* __restrict__ Wq, const float* __restrict__ bq,
    const float* __restrict__ Wk, const float* __restrict__ bk,
    const float* __restrict__ Wv, const float* __restrict__ bv,
    _Float16* __restrict__ Qh, _Float16* __restrict__ Kh,
    _Float16* __restrict__ Vt) {
  __shared__ __align__(16) _Float16 ldsA[128 * 40];  // [row][k], pad 40
  __shared__ __align__(16) _Float16 ldsB[64 * 40];   // transposed: [n][k]

  const int t = threadIdx.x;
  const int z = blockIdx.z;
  const float* W  = (z == 0) ? Wq : (z == 1) ? Wk : Wv;
  const float* bi = (z == 0) ? bq : (z == 1) ? bk : bv;
  _Float16*   Out = (z == 0) ? Qh : Kh;

  const int rowBase = blockIdx.x * 128;
  const int colBase = blockIdx.y * 64;

  const int wave = t >> 5, lane = t & 31;
  const int g = lane >> 4, m = lane & 15;
  const int waveM = wave >> 1, waveN = wave & 1;

  const int rowA = t >> 1;          // 0..127
  const int colA = (t & 1) * 16;    // 0 or 16
  const int iB = t * 8;             // over 32k x 64n W tile
  const int kB = iB >> 6;           // 0..31
  const int nB = iB & 63;           // multiple of 8

  v8f acc[2][2] = {};

  for (int kk = 0; kk < D_; kk += 32) {
    __syncthreads();
    {  // A tile: x[rowBase+rowA][kk+colA .. +15] -> f16
      const float* gp = x + (size_t)(rowBase + rowA) * D_ + kk + colA;
      float4 f0 = ((const float4*)gp)[0];
      float4 f1 = ((const float4*)gp)[1];
      float4 f2 = ((const float4*)gp)[2];
      float4 f3 = ((const float4*)gp)[3];
      half8 h0 = {(_Float16)f0.x, (_Float16)f0.y, (_Float16)f0.z, (_Float16)f0.w,
                  (_Float16)f1.x, (_Float16)f1.y, (_Float16)f1.z, (_Float16)f1.w};
      half8 h1 = {(_Float16)f2.x, (_Float16)f2.y, (_Float16)f2.z, (_Float16)f2.w,
                  (_Float16)f3.x, (_Float16)f3.y, (_Float16)f3.z, (_Float16)f3.w};
      *(half8*)&ldsA[rowA * 40 + colA]     = h0;
      *(half8*)&ldsA[rowA * 40 + colA + 8] = h1;
      if (kk + 32 < D_) __builtin_prefetch(gp + 32, 0, 1);
    }
    {  // W tile: coalesced read, store transposed [n][k]
      const float* wp = W + (size_t)(kk + kB) * D_ + colBase + nB;
      float4 w0 = ((const float4*)wp)[0];
      float4 w1 = ((const float4*)wp)[1];
      float vals[8] = {w0.x, w0.y, w0.z, w0.w, w1.x, w1.y, w1.z, w1.w};
#pragma unroll
      for (int j = 0; j < 8; ++j)
        ldsB[(nB + j) * 40 + kB] = (_Float16)vals[j];
      if (kk + 32 < D_) __builtin_prefetch(wp + (size_t)32 * D_, 0, 1);
    }
    __syncthreads();

    v16h a0 = load_fragA(&ldsA[(waveM * 32 +      m) * 40], g, 0);
    v16h a1 = load_fragA(&ldsA[(waveM * 32 + 16 + m) * 40], g, 0);
    v16h b0 = load_fragB(&ldsB[(waveN * 32 +      m) * 40], g, 0);
    v16h b1 = load_fragB(&ldsB[(waveN * 32 + 16 + m) * 40], g, 0);

    acc[0][0] = wmma32(a0, b0, acc[0][0]);
    acc[0][1] = wmma32(a0, b1, acc[0][1]);
    acc[1][0] = wmma32(a1, b0, acc[1][0]);
    acc[1][1] = wmma32(a1, b1, acc[1][1]);
  }

  const float LT = 9.210340371976184f / 32.0f;  // ln(10000)/(Dh/2)
#pragma unroll
  for (int i = 0; i < 2; ++i) {
#pragma unroll
    for (int j = 0; j < 2; ++j) {
      const int col  = colBase + waveN * 32 + j * 16 + m;
      const float bcol = bi[col];
      const int head = col >> 6;
      const int dcol = col & 63;
      const float invf = __expf(-(float)(dcol >> 1) * LT);
#pragma unroll
      for (int r = 0; r < 8; ++r) {
        const int row = rowBase + waveM * 32 + i * 16 + r + 8 * g;
        const int bb = row >> 11;        // / N
        const int n  = row & (N_ - 1);   // % N
        float val = acc[i][j][r] + bcol;
        if (z == 2) {  // V: transposed [B,H,Dh,N]
          Vt[(((size_t)bb * H_ + head) * DH_ + dcol) * N_ + n] = (_Float16)val;
        } else {       // Q,K: RoPE; partner column lives in lane^1 of C layout
          float partner = __shfl_xor(val, 1);
          float ang = (float)n * invf;
          float sn, cs;
          __sincosf(ang, &sn, &cs);
          float rot = (col & 1) ? partner : -partner;
          val = val * cs + rot * sn;
          Out[(((size_t)bb * H_ + head) * N_ + n) * DH_ + dcol] = (_Float16)val;
        }
      }
    }
  }
}

// ---------------------------------------------------------------------------
// Kernel 2: flash attention.  grid = (N/128, H, B), block = 256.
// Each wave owns 16 query rows.  K/V stream through double-buffered LDS via
// the Tensor Data Mover (wave 0 issues; TENSORcnt + barriers synchronize).
// TDM pad_enable produces the 72/40-half row strides the fragment loaders use.
// ---------------------------------------------------------------------------
__global__ __launch_bounds__(256) void flash_attn(
    const _Float16* __restrict__ Qh, const _Float16* __restrict__ Kh,
    const _Float16* __restrict__ Vt, _Float16* __restrict__ inter) {
  __shared__ __align__(16) _Float16 ldsK[2][32 * 72];   // [key][dh]
  __shared__ __align__(16) _Float16 ldsV[2][64 * 40];   // [dh][key]
  __shared__ __align__(16) _Float16 ldsP[8 * 16 * 40];  // per-wave P patch

  const int t = threadIdx.x;
  const int wave = t >> 5, lane = t & 31;
  const int g = lane >> 4, m = lane & 15;
  const int h = blockIdx.y, b = blockIdx.z;
  const int bh = b * H_ + h;
  const int qBase = blockIdx.x * 128;

  const _Float16* Kbase = Kh + (size_t)bh * N_ * DH_;  // [n][dh]
  const _Float16* Vbase = Vt + (size_t)bh * DH_ * N_;  // [dh][n]

  // Persistent Q strip (16 rows x 64) as two A fragments, straight from global.
  const _Float16* qrow = Qh + ((size_t)bh * N_ + qBase + wave * 16 + m) * DH_;
  v16h aQ0 = load_fragA(qrow, g, 0);
  v16h aQ1 = load_fragA(qrow, g, 32);

  float mi[8], li[8], alpha[8];
  v8f o[4] = {};
#pragma unroll
  for (int r = 0; r < 8; ++r) { mi[r] = -1e30f; li[r] = 0.0f; }

  const float SCALE = 0.03125f;  // reference scales by 1/sqrt(D) = 1/32

#if HAVE_TDM
  if (wave == 0) {  // prologue: DMA tile 0 into buffer 0
    tdm_load_2d_f16((unsigned int)(uintptr_t)&ldsK[0][0], Kbase, PAD_ROW64H,
                    DH_, N_, DH_, 32, DH_);
    tdm_load_2d_f16((unsigned int)(uintptr_t)&ldsV[0][0], Vbase, PAD_ROW32H,
                    N_, DH_, 32, DH_, N_);
  }
#else
  const int i8 = t * 8;
  const int kRow = i8 >> 6, dh0 = i8 & 63;      // K tile copy: 32x64
  const int vRow = t >> 2, vCol = (t & 3) * 8;  // V tile copy: 64x32
#endif

  int buf = 0;
  for (int kb = 0; kb < N_; kb += 32, buf ^= 1) {
    __syncthreads();  // all reads of buf^1 retired -> safe to refill it
#if HAVE_TDM
    if (wave == 0) {
      if (kb + 32 < N_) {  // DMA next tile into the other buffer
        tdm_load_2d_f16((unsigned int)(uintptr_t)&ldsK[buf ^ 1][0],
                        Kbase + (size_t)(kb + 32) * DH_, PAD_ROW64H,
                        DH_, N_, DH_, 32, DH_);
        tdm_load_2d_f16((unsigned int)(uintptr_t)&ldsV[buf ^ 1][0],
                        Vbase + (kb + 32), PAD_ROW32H,
                        N_, DH_, 32, DH_, N_);
        __builtin_amdgcn_s_wait_tensorcnt(2);  // current tile complete
      } else {
        __builtin_amdgcn_s_wait_tensorcnt(0);
      }
    }
#else
    {
      const _Float16* gk = Kbase + (size_t)(kb + kRow) * DH_ + dh0;
      *(half8*)&ldsK[buf][kRow * 72 + dh0] = *(const half8*)gk;
      const _Float16* gv = Vbase + (size_t)vRow * N_ + kb + vCol;
      *(half8*)&ldsV[buf][vRow * 40 + vCol] = *(const half8*)gv;
    }
#endif
    __syncthreads();  // buf is ready for everyone

    // S = Q . K^T  (16 rows x 32 keys): 2 key tiles x 2 K-chunks of 32
    v8f s0 = {}, s1 = {};
    {
      v16h bk00 = load_fragB(&ldsK[buf][(0 + m) * 72], g, 0);
      v16h bk01 = load_fragB(&ldsK[buf][(0 + m) * 72], g, 32);
      v16h bk10 = load_fragB(&ldsK[buf][(16 + m) * 72], g, 0);
      v16h bk11 = load_fragB(&ldsK[buf][(16 + m) * 72], g, 32);
      s0 = wmma32(aQ0, bk00, s0);
      s0 = wmma32(aQ1, bk01, s0);
      s1 = wmma32(aQ0, bk10, s1);
      s1 = wmma32(aQ1, bk11, s1);
    }

    // Online softmax; xor-1/2/4/8 butterflies reduce over a row's 16 lanes.
#pragma unroll
    for (int r = 0; r < 8; ++r) {
      float v0 = s0[r] * SCALE, v1 = s1[r] * SCALE;
      float mx = fmaxf(v0, v1);
#pragma unroll
      for (int off = 1; off < 16; off <<= 1) mx = fmaxf(mx, __shfl_xor(mx, off));
      float mn = fmaxf(mi[r], mx);
      float p0 = __expf(v0 - mn);
      float p1 = __expf(v1 - mn);
      float rs = p0 + p1;
#pragma unroll
      for (int off = 1; off < 16; off <<= 1) rs += __shfl_xor(rs, off);
      alpha[r] = __expf(mi[r] - mn);
      li[r] = li[r] * alpha[r] + rs;
      mi[r] = mn;
      s0[r] = p0;
      s1[r] = p1;
    }
#pragma unroll
    for (int tt = 0; tt < 4; ++tt)
#pragma unroll
      for (int r = 0; r < 8; ++r) o[tt][r] *= alpha[r];

    // Re-layout P (C-frag) -> A-frag via per-wave LDS patch (same-wave LDS
    // ops are in order; no barrier needed).
    {
      _Float16* pP = &ldsP[wave * 16 * 40];
#pragma unroll
      for (int r = 0; r < 8; ++r) {
        const int row = r + 8 * g;
        pP[row * 40 + m]      = (_Float16)s0[r];
        pP[row * 40 + 16 + m] = (_Float16)s1[r];
      }
    }
    v16h aP = load_fragA(&ldsP[(wave * 16 + m) * 40], g, 0);
#pragma unroll
    for (int tt = 0; tt < 4; ++tt) {
      v16h bv = load_fragB(&ldsV[buf][(tt * 16 + m) * 40], g, 0);
      o[tt] = wmma32(aP, bv, o[tt]);
    }
  }

  // O = o / l, store f16 into inter[b, n, h*64+dh]
#pragma unroll
  for (int tt = 0; tt < 4; ++tt) {
#pragma unroll
    for (int r = 0; r < 8; ++r) {
      const int row = qBase + wave * 16 + r + 8 * g;
      const int col = h * DH_ + tt * 16 + m;
      inter[((size_t)b * N_ + row) * D_ + col] = (_Float16)(o[tt][r] / li[r]);
    }
  }
}

// ---------------------------------------------------------------------------
// Kernel 3: output projection, f16 A @ f16(Wo) + bo -> f32 out.
// grid = (M/128, D/64), block = 256.
// ---------------------------------------------------------------------------
__global__ __launch_bounds__(256) void out_proj_gemm(
    const _Float16* __restrict__ A, const float* __restrict__ Wo,
    const float* __restrict__ bo, float* __restrict__ out) {
  __shared__ __align__(16) _Float16 ldsA[128 * 40];
  __shared__ __align__(16) _Float16 ldsB[64 * 40];

  const int t = threadIdx.x;
  const int rowBase = blockIdx.x * 128;
  const int colBase = blockIdx.y * 64;
  const int wave = t >> 5, lane = t & 31;
  const int g = lane >> 4, m = lane & 15;
  const int waveM = wave >> 1, waveN = wave & 1;

  const int rowA = t >> 1;
  const int colA = (t & 1) * 16;
  const int iB = t * 8;
  const int kB = iB >> 6;
  const int nB = iB & 63;

  v8f acc[2][2] = {};

  for (int kk = 0; kk < D_; kk += 32) {
    __syncthreads();
    {  // A tile already f16: two 16B copies
      const _Float16* gp = A + (size_t)(rowBase + rowA) * D_ + kk + colA;
      *(half8*)&ldsA[rowA * 40 + colA]     = ((const half8*)gp)[0];
      *(half8*)&ldsA[rowA * 40 + colA + 8] = ((const half8*)gp)[1];
      if (kk + 32 < D_) __builtin_prefetch(gp + 32, 0, 1);
    }
    {
      const float* wp = Wo + (size_t)(kk + kB) * D_ + colBase + nB;
      float4 w0 = ((const float4*)wp)[0];
      float4 w1 = ((const float4*)wp)[1];
      float vals[8] = {w0.x, w0.y, w0.z, w0.w, w1.x, w1.y, w1.z, w1.w};
#pragma unroll
      for (int j = 0; j < 8; ++j)
        ldsB[(nB + j) * 40 + kB] = (_Float16)vals[j];
      if (kk + 32 < D_) __builtin_prefetch(wp + (size_t)32 * D_, 0, 1);
    }
    __syncthreads();

    v16h a0 = load_fragA(&ldsA[(waveM * 32 +      m) * 40], g, 0);
    v16h a1 = load_fragA(&ldsA[(waveM * 32 + 16 + m) * 40], g, 0);
    v16h b0 = load_fragB(&ldsB[(waveN * 32 +      m) * 40], g, 0);
    v16h b1 = load_fragB(&ldsB[(waveN * 32 + 16 + m) * 40], g, 0);

    acc[0][0] = wmma32(a0, b0, acc[0][0]);
    acc[0][1] = wmma32(a0, b1, acc[0][1]);
    acc[1][0] = wmma32(a1, b0, acc[1][0]);
    acc[1][1] = wmma32(a1, b1, acc[1][1]);
  }

#pragma unroll
  for (int i = 0; i < 2; ++i) {
#pragma unroll
    for (int j = 0; j < 2; ++j) {
      const int col = colBase + waveN * 32 + j * 16 + m;
      const float bcol = bo[col];
#pragma unroll
      for (int r = 0; r < 8; ++r) {
        const int row = rowBase + waveM * 32 + i * 16 + r + 8 * g;
        out[(size_t)row * D_ + col] = acc[i][j][r] + bcol;
      }
    }
  }
}

// ---------------------------------------------------------------------------
extern "C" void kernel_launch(void* const* d_in, const int* in_sizes, int n_in,
                              void* d_out, int out_size, void* d_ws,
                              size_t ws_size, hipStream_t stream) {
  const float* x  = (const float*)d_in[0];
  const float* Wq = (const float*)d_in[1];
  const float* bq = (const float*)d_in[2];
  const float* Wk = (const float*)d_in[3];
  const float* bk = (const float*)d_in[4];
  const float* Wv = (const float*)d_in[5];
  const float* bv = (const float*)d_in[6];
  const float* Wo = (const float*)d_in[7];
  const float* bo = (const float*)d_in[8];
  float* out = (float*)d_out;

  // Workspace layout (f16): Qh | Kh | Vt | inter, each B*H*N*Dh = 8M halves.
  const size_t SEG = (size_t)B_ * H_ * N_ * DH_;
  if (ws_size < 4 * SEG * sizeof(_Float16)) return;  // need 64 MB scratch
  _Float16* Qh    = (_Float16*)d_ws;
  _Float16* Kh    = Qh + SEG;
  _Float16* Vt    = Kh + SEG;
  _Float16* inter = Vt + SEG;

  dim3 blk(256, 1, 1);
  qkv_rope_gemm<<<dim3((B_ * N_) / 128, D_ / 64, 3), blk, 0, stream>>>(
      x, Wq, bq, Wk, bk, Wv, bv, Qh, Kh, Vt);
  flash_attn<<<dim3(N_ / 128, H_, B_), blk, 0, stream>>>(Qh, Kh, Vt, inter);
  out_proj_gemm<<<dim3((B_ * N_) / 128, D_ / 64, 1), blk, 0, stream>>>(
      inter, Wo, bo, out);
}
